// GRUv2_3229815407362
// MI455X (gfx1250) — matrix operand order repaired and checked by reading, not compiled, not run on hardware
//
#include <hip/hip_runtime.h>
#include <hip/hip_bf16.h>

typedef __attribute__((ext_vector_type(16))) __bf16 v16bf;
typedef __attribute__((ext_vector_type(8)))  float  v8f;

constexpr int kH = 1024;   // hidden
constexpr int kL = 4;      // layers
constexpr int kT = 4096;   // seq len
constexpr int kTPB = 256;  // threads per block (8 waves, wave32)
constexpr int kNBLK = 32;  // persistent grid
constexpr int kNWAVE = kNBLK * (kTPB / 32);  // 256 persistent waves

union Frag {
  uint4 q[2];
  v16bf v;
};

__device__ __forceinline__ unsigned short f2bf(float f) {
  unsigned u = __float_as_uint(f);
  u += 0x7FFFu + ((u >> 16) & 1u);  // round-to-nearest-even
  return (unsigned short)(u >> 16);
}

__device__ __forceinline__ float sigm(float x) { return 1.0f / (1.0f + __expf(-x)); }

#define WMMA_BF16(A, B, C) \
  __builtin_amdgcn_wmma_f32_16x16x32_bf16(false, (A).v, false, (B).v, (short)0, (C), false, false)

// ---------------------------------------------------------------------------
// fp32 -> bf16 pack
// ---------------------------------------------------------------------------
__global__ void __launch_bounds__(256) cvt_bf16(const float* __restrict__ src,
                                               unsigned short* __restrict__ dst, int n) {
  int i = blockIdx.x * blockDim.x + threadIdx.x;
  if (i < n) dst[i] = f2bf(src[i]);
}

// ---------------------------------------------------------------------------
// state + barrier init (ws is poisoned with 0xAA by the harness)
// ---------------------------------------------------------------------------
__global__ void __launch_bounds__(256) init_state(float* __restrict__ h,
                                                  unsigned short* __restrict__ hb,
                                                  unsigned* __restrict__ bar) {
  int i = blockIdx.x * blockDim.x + threadIdx.x;
  if (i < kL * kH) { h[i] = 0.0f; hb[i] = 0; }
  if (i < 128) bar[i] = 0u;
}

// ---------------------------------------------------------------------------
// Input projections: P[g][t][m] = sum_k Ug[m,k] * x[t,k], g in {u,r,c}
// One wave -> 16(t) x 64(m) tile, K blocked by 32, 2-stage SW pipeline so the
// s_wait_loadcnt before each WMMA leaves the next stage's loads in flight.
// ---------------------------------------------------------------------------
__global__ void __launch_bounds__(kTPB, 1) proj_gemm(
    const unsigned short* __restrict__ xb, const unsigned short* __restrict__ Uu,
    const unsigned short* __restrict__ Ur, const unsigned short* __restrict__ Uc,
    float* __restrict__ P) {
  const int lane = threadIdx.x & 31;
  const int wid  = threadIdx.x >> 5;
  const int task = blockIdx.x * (kTPB / 32) + wid;  // 0..12287
  const int proj = task >> 12;                      // 4096 tiles per projection
  const int rem  = task & 4095;
  const int t0 = (rem >> 4) << 4;                   // 256 t-tiles
  const int m0 = (rem & 15) << 6;                   // 16 m-blocks of 64

  const unsigned short* Ug = (proj == 0) ? Uu : (proj == 1) ? Ur : Uc;
  float* Pg = P + (size_t)proj * kT * kH;

  const int sub = lane & 15, hi = lane >> 4;
  // A fragment: row t0+sub, K chunks {0..7,16..23} (lo lanes) / {8..15,24..31} (hi lanes)
  const unsigned short* xr = xb + (size_t)(t0 + sub) * kH + (hi ? 8 : 0);
  // B fragment: col m0+sub, K chunks {0..7,16..23} (lo lanes) / {8..15,24..31} (hi lanes)
  const unsigned short* ub = Ug + (size_t)(m0 + sub) * kH + (hi ? 8 : 0);

  Frag a0, a1, b0[4], b1[4];
#define LOAD_STAGE(A, B, K)                                      \
  do {                                                           \
    (A).q[0] = *(const uint4*)(xr + (K));                        \
    (A).q[1] = *(const uint4*)(xr + (K) + 16);                   \
    _Pragma("unroll") for (int j = 0; j < 4; ++j) {              \
      (B)[j].q[0] = *(const uint4*)(ub + j * 16 * kH + (K));     \
      (B)[j].q[1] = *(const uint4*)(ub + j * 16 * kH + (K) + 16); \
    }                                                            \
  } while (0)

  v8f c0 = {}, c1 = {}, c2 = {}, c3 = {};
  LOAD_STAGE(a0, b0, 0);
  for (int k0 = 0; k0 < kH; k0 += 64) {
    LOAD_STAGE(a1, b1, k0 + 32);  // prefetch next K-slab while stage0 computes
    c0 = WMMA_BF16(a0, b0[0], c0);
    c1 = WMMA_BF16(a0, b0[1], c1);
    c2 = WMMA_BF16(a0, b0[2], c2);
    c3 = WMMA_BF16(a0, b0[3], c3);
    if (k0 + 64 < kH) LOAD_STAGE(a0, b0, k0 + 64);
    c0 = WMMA_BF16(a1, b1[0], c0);
    c1 = WMMA_BF16(a1, b1[1], c1);
    c2 = WMMA_BF16(a1, b1[2], c2);
    c3 = WMMA_BF16(a1, b1[3], c3);
  }
#undef LOAD_STAGE

  // D: lanes 0-15 -> rows t0..t0+7, lanes 16-31 -> rows t0+8..t0+15; col = m0+sub+16j
  const int col = m0 + sub;
  const int rb = t0 + (hi ? 8 : 0);
#pragma unroll
  for (int r = 0; r < 8; ++r) {
    Pg[(size_t)(rb + r) * kH + col + 0]  = c0[r];
    Pg[(size_t)(rb + r) * kH + col + 16] = c1[r];
    Pg[(size_t)(rb + r) * kH + col + 32] = c2[r];
    Pg[(size_t)(rb + r) * kH + col + 48] = c3[r];
  }
}

// ---------------------------------------------------------------------------
// WMMA mat-vec: out[m0..m0+15] = W[m0..m0+15,:] @ v. One wave; v sits in B
// column 0 (lanes 0/16). A-tile stream from L2 is double-buffered.
// ---------------------------------------------------------------------------
__device__ __forceinline__ void matvec16(const unsigned short* __restrict__ w,  // HxH bf16
                                         const unsigned short* v,               // LDS, kH bf16
                                         float* __restrict__ out, int m0, int lane) {
  v8f acc = {};
  const int sub = lane & 15, hi = lane >> 4;
  const unsigned short* wr = w + (size_t)(m0 + sub) * kH + (hi ? 8 : 0);
  const bool bAct = (sub == 0);  // column 0 lives in lanes 0 and 16
  const unsigned short* vr = v + (hi ? 8 : 0);

  Frag a0, a1;
  a0.q[0] = *(const uint4*)(wr + 0);
  a0.q[1] = *(const uint4*)(wr + 16);
  for (int k0 = 0; k0 < kH; k0 += 64) {
    a1.q[0] = *(const uint4*)(wr + k0 + 32);
    a1.q[1] = *(const uint4*)(wr + k0 + 48);
    Frag b;
    if (bAct) {
      b.q[0] = *(const uint4*)(vr + k0);
      b.q[1] = *(const uint4*)(vr + k0 + 16);
    } else {
      b.q[0] = make_uint4(0u, 0u, 0u, 0u);
      b.q[1] = make_uint4(0u, 0u, 0u, 0u);
    }
    acc = WMMA_BF16(a0, b, acc);
    if (k0 + 64 < kH) {
      a0.q[0] = *(const uint4*)(wr + k0 + 64);
      a0.q[1] = *(const uint4*)(wr + k0 + 80);
    }
    if (bAct) {
      b.q[0] = *(const uint4*)(vr + k0 + 32);
      b.q[1] = *(const uint4*)(vr + k0 + 48);
    }
    acc = WMMA_BF16(a1, b, acc);
  }
  if (bAct) {
    float* o = out + m0 + (hi ? 8 : 0);
#pragma unroll
    for (int r = 0; r < 8; ++r) o[r] = acc[r];
  }
}

// ---------------------------------------------------------------------------
// Async global -> LDS staging (CDNA5 GLOBAL_LOAD_ASYNC_TO_LDS_B128, ASYNCcnt).
// Generic pointer low 32 bits == wave-relative LDS byte address (aperture rule).
// ---------------------------------------------------------------------------
__device__ __forceinline__ void async_stage(const unsigned short* __restrict__ g,
                                            unsigned short* l, int nchunk16) {
  const unsigned ldsBase = (unsigned)(uintptr_t)l;
  for (int i = threadIdx.x; i < nchunk16; i += kTPB) {
    unsigned lds = ldsBase + (unsigned)i * 16u;
    const void* ga = (const void*)((const char*)g + (size_t)i * 16u);
    asm volatile("global_load_async_to_lds_b128 %0, %1, off" ::"v"(lds), "v"(ga) : "memory");
  }
  asm volatile("s_wait_asynccnt 0x0" ::: "memory");
}

// ---------------------------------------------------------------------------
// Grid-wide split barrier (persistent kernel, agent scope)
// ---------------------------------------------------------------------------
__device__ __forceinline__ void grid_barrier(unsigned* bar) {
  __syncthreads();
  if (threadIdx.x == 0) {
    unsigned* cnt = bar;
    unsigned* gen = bar + 64;  // separate cacheline
    unsigned g = __hip_atomic_load(gen, __ATOMIC_RELAXED, __HIP_MEMORY_SCOPE_AGENT);
    unsigned a = __hip_atomic_fetch_add(cnt, 1u, __ATOMIC_ACQ_REL, __HIP_MEMORY_SCOPE_AGENT);
    if (a == (unsigned)(kNBLK - 1)) {
      __hip_atomic_store(cnt, 0u, __ATOMIC_RELAXED, __HIP_MEMORY_SCOPE_AGENT);
      __hip_atomic_store(gen, g + 1u, __ATOMIC_RELEASE, __HIP_MEMORY_SCOPE_AGENT);
    } else {
      while (__hip_atomic_load(gen, __ATOMIC_ACQUIRE, __HIP_MEMORY_SCOPE_AGENT) == g) {
        __builtin_amdgcn_s_sleep(8);
      }
    }
  }
  __syncthreads();
}

// ---------------------------------------------------------------------------
// Persistent GRU scan. Weights (~42 MB bf16) stay L2-resident on MI455X.
// Per timestep: phase A (12 parallel W@h matvecs), then per layer l:
//   [l>0] Uh/Uhr @ hnew[l-1], then elementwise gates + state update.
// ---------------------------------------------------------------------------
__global__ void __launch_bounds__(kTPB, 1) gru_persist(
    const unsigned short* __restrict__ WuB, const unsigned short* __restrict__ WrB,
    const unsigned short* __restrict__ WcB, const unsigned short* __restrict__ UhrB,
    const unsigned short* __restrict__ UhB, const float* __restrict__ P,
    const float* __restrict__ Bu, const float* __restrict__ Br, const float* __restrict__ Bc,
    float* __restrict__ h, unsigned short* __restrict__ hb, float* __restrict__ gW,
    float* __restrict__ gU, unsigned* bar, float* __restrict__ out) {
  __shared__ __align__(16) unsigned short s_vec[kL * kH];
  const int lane  = threadIdx.x & 31;
  const int wid   = threadIdx.x >> 5;
  const int gwave = blockIdx.x * (kTPB / 32) + wid;
  const int gtid  = blockIdx.x * kTPB + threadIdx.x;

  for (int t = 0; t < kT; ++t) {
    // Stage full hidden stack [kL*kH] bf16 into LDS (8 KB) via async DMA
    async_stage(hb, s_vec, (kL * kH) / 8);
    __syncthreads();

    // Phase A: au/ar/ac[l] = {Wu,Wr,W}[l] @ h[l]   (12 matvecs, 768 tiles)
    for (int task = gwave; task < 12 * 64; task += kNWAVE) {
      const int g = task >> 6, m0 = (task & 63) << 4;
      const int l = g / 3, which = g % 3;
      const unsigned short* Wm =
          ((which == 0) ? WuB : (which == 1) ? WrB : WcB) + (size_t)l * kH * kH;
      matvec16(Wm, s_vec + l * kH, gW + g * kH, m0, lane);
    }
    grid_barrier(bar);

    for (int l = 0; l < kL; ++l) {
      if (l > 0) {
        // Stage hnew[l-1] into LDS, then shared = Uh[l-1]@prev, ir = Uhr[l-1]@prev
        async_stage(hb + (size_t)(l - 1) * kH, s_vec, kH / 8);
        __syncthreads();
        for (int task = gwave; task < 2 * 64; task += kNWAVE) {
          const int g = task >> 6, m0 = (task & 63) << 4;
          const unsigned short* Um = ((g == 0) ? UhB : UhrB) + (size_t)(l - 1) * kH * kH;
          matvec16(Um, s_vec, gU + g * kH, m0, lane);
        }
        grid_barrier(bar);
      }
      // Elementwise gate math + state update (1024 lanes of work)
      if (gtid < kH) {
        const int i = gtid;
        float iu, ir, ic;
        if (l == 0) {
          iu = P[(size_t)t * kH + i];
          ir = P[(size_t)kT * kH + (size_t)t * kH + i];
          ic = P[2 * (size_t)kT * kH + (size_t)t * kH + i];
        } else {
          iu = gU[i];       // Uh[l-1]@prev shared by update gate and candidate
          ic = iu;
          ir = gU[kH + i];  // Uhr[l-1]@prev
        }
        const float u = sigm(iu + gW[(l * 3 + 0) * kH + i] + Bu[l * kH + i]);
        const float r = sigm(ir + gW[(l * 3 + 1) * kH + i] + Br[l * kH + i]);
        const float c = sigm(ic + r * gW[(l * 3 + 2) * kH + i] + Bc[l * kH + i]);
        const float hp = h[l * kH + i];
        const float hn = u * hp + (1.0f - u) * c;
        h[l * kH + i]  = hn;
        hb[l * kH + i] = f2bf(hn);
      }
      grid_barrier(bar);
    }
  }
  if (gtid < kL * kH) out[gtid] = h[gtid];
}

// ---------------------------------------------------------------------------
extern "C" void kernel_launch(void* const* d_in, const int* in_sizes, int n_in,
                              void* d_out, int out_size, void* d_ws, size_t ws_size,
                              hipStream_t stream) {
  const float* x   = (const float*)d_in[0];
  const float* Uu  = (const float*)d_in[1];
  const float* Ur  = (const float*)d_in[2];
  const float* U   = (const float*)d_in[3];
  const float* Wu  = (const float*)d_in[4];
  const float* Wr  = (const float*)d_in[5];
  const float* W   = (const float*)d_in[6];
  const float* Bu  = (const float*)d_in[7];
  const float* Br  = (const float*)d_in[8];
  const float* B   = (const float*)d_in[9];
  const float* Uhr = (const float*)d_in[10];
  const float* Uh  = (const float*)d_in[11];

  char* ws = (char*)d_ws;
  size_t off = 0;
  auto alloc = [&](size_t bytes) {
    char* p = ws + off;
    off += (bytes + 255) & ~(size_t)255;
    return p;
  };
  unsigned short* xb   = (unsigned short*)alloc((size_t)kT * kH * 2);
  unsigned short* UuB  = (unsigned short*)alloc((size_t)kH * kH * 2);
  unsigned short* UrB  = (unsigned short*)alloc((size_t)kH * kH * 2);
  unsigned short* UcB  = (unsigned short*)alloc((size_t)kH * kH * 2);
  unsigned short* WuB  = (unsigned short*)alloc((size_t)kL * kH * kH * 2);
  unsigned short* WrB  = (unsigned short*)alloc((size_t)kL * kH * kH * 2);
  unsigned short* WcB  = (unsigned short*)alloc((size_t)kL * kH * kH * 2);
  unsigned short* UhrB = (unsigned short*)alloc((size_t)(kL - 1) * kH * kH * 2);
  unsigned short* UhB  = (unsigned short*)alloc((size_t)(kL - 1) * kH * kH * 2);
  float* P  = (float*)alloc((size_t)3 * kT * kH * 4);
  float* h  = (float*)alloc((size_t)kL * kH * 4);
  unsigned short* hb = (unsigned short*)alloc((size_t)kL * kH * 2);
  float* gW = (float*)alloc((size_t)12 * kH * 4);
  float* gU = (float*)alloc((size_t)2 * kH * 4);
  unsigned* bar = (unsigned*)alloc(512);

  auto cvt = [&](const float* s, unsigned short* d, int n) {
    cvt_bf16<<<(n + 255) / 256, 256, 0, stream>>>(s, d, n);
  };
  cvt(x, xb, kT * kH);
  cvt(Uu, UuB, kH * kH);
  cvt(Ur, UrB, kH * kH);
  cvt(U, UcB, kH * kH);
  cvt(Wu, WuB, kL * kH * kH);
  cvt(Wr, WrB, kL * kH * kH);
  cvt(W, WcB, kL * kH * kH);
  cvt(Uhr, UhrB, (kL - 1) * kH * kH);
  cvt(Uh, UhB, (kL - 1) * kH * kH);

  // 3 projections x 4096 wave-tiles, 8 waves/block -> 1536 blocks
  proj_gemm<<<1536, kTPB, 0, stream>>>(xb, UuB, UrB, UcB, P);

  init_state<<<(kL * kH + 255) / 256, 256, 0, stream>>>(h, hb, bar);

  gru_persist<<<kNBLK, kTPB, 0, stream>>>(WuB, WrB, WcB, UhrB, UhB, P, Bu, Br, B, h, hb,
                                          gW, gU, bar, (float*)d_out);
}
